// LlamaTransformerLayer_8632884265312
// MI455X (gfx1250) — compile-verified, matrix operand order checked by reading, not run on hardware
//
#include <hip/hip_runtime.h>
#include <hip/hip_bf16.h>

typedef __bf16 bf16_t;
typedef __attribute__((ext_vector_type(16))) __bf16 v16bf;
typedef __attribute__((ext_vector_type(8)))  __bf16 v8bf;
typedef __attribute__((ext_vector_type(8)))  float   v8f;
typedef __attribute__((ext_vector_type(4)))  float   v4f;

union V16U { v16bf v; v8bf h[2]; };

// Load one 16x16(x32) bf16 WMMA fragment: halves i=0..7 -> K+0..7, i=8..15 -> K+16..23
static __device__ __forceinline__ v16bf ld_frag(const bf16_t* base) {
  V16U u;
  u.h[0] = *(const v8bf*)(base);
  u.h[1] = *(const v8bf*)(base + 16);
  return u.v;
}

// Wave-relative LDS byte offset: for generic pointers in the LDS aperture,
// addr[31:0] is the LDS offset (ISA 10.2 aperture mapping).
static __device__ __forceinline__ unsigned lds_off(const void* p) {
  return (unsigned)(uintptr_t)p;
}

// CDNA5 async copy: 16B global -> LDS, tracked by ASYNCcnt (ISA 15.18.3 op 98).
static __device__ __forceinline__ void async_cp16(unsigned lds_byte, const void* gsrc) {
  asm volatile("global_load_async_to_lds_b128 %0, %1, off"
               :: "v"(lds_byte), "v"(gsrc) : "memory");
}
static __device__ __forceinline__ void wait_async0() {
  asm volatile("s_wait_asynccnt 0x0" ::: "memory");
}

static __device__ __forceinline__ unsigned f2ord(float f) {
  unsigned u = __float_as_uint(f);
  return (u & 0x80000000u) ? ~u : (u | 0x80000000u);
}
static __device__ __forceinline__ float ord2f(unsigned u) {
  u = (u & 0x80000000u) ? (u ^ 0x80000000u) : ~u;
  return __uint_as_float(u);
}

// ---------------------------------------------------------------- RMSNorm
__global__ __launch_bounds__(256) void rmsnorm_kernel(
    const float* __restrict__ x, const float* __restrict__ g,
    bf16_t* __restrict__ out, int D)
{
  __shared__ float red[256];
  const int row = blockIdx.x;
  const float* xr = x + (size_t)row * D;
  float s = 0.f;
  for (int i = threadIdx.x; i < D; i += 256) { float v = xr[i]; s += v * v; }
  red[threadIdx.x] = s;
  __syncthreads();
  for (int off = 128; off > 0; off >>= 1) {
    if (threadIdx.x < off) red[threadIdx.x] += red[threadIdx.x + off];
    __syncthreads();
  }
  const float scale = rsqrtf(red[0] / (float)D + 1e-6f);
  bf16_t* orow = out + (size_t)row * D;
  for (int i = threadIdx.x; i < D; i += 256)
    orow[i] = (bf16_t)(xr[i] * scale * g[i]);
}

// ---------------------------------------------------------------- GEMM C[M,N] = A[M,K](bf16) * W[N,K]^T(fp32) (+RES)
#define GLDA 40  // halves per LDS row (32 data + 8 pad): 16B-aligned rows, conflict-free frag loads

__global__ __launch_bounds__(256) void gemm_bf16_kernel(
    const bf16_t* __restrict__ A, const float* __restrict__ W,
    const float* __restrict__ RES, float* __restrict__ C,
    int M, int N, int K, int epi)
{
  __shared__ bf16_t sA[2][128 * GLDA];
  __shared__ bf16_t sB[2][128 * GLDA];

  const int t    = threadIdx.x;
  const int bm   = blockIdx.y * 128;
  const int bn   = blockIdx.x * 128;
  const int lane = t & 31;
  const int wid  = t >> 5;
  const int wm   = (wid & 3) * 32;   // 4 waves over M
  const int wn   = (wid >> 2) * 64;  // 2 waves over N
  const int fr   = lane & 15;
  const int fk   = (lane >> 4) * 8;

  const int srow = t >> 1;
  const int scol = (t & 1) * 16;

  v8f acc[2][4];
#pragma unroll
  for (int i = 0; i < 2; ++i)
#pragma unroll
    for (int j = 0; j < 4; ++j) acc[i][j] = (v8f)0.f;

  const int KB = K >> 5;

  auto stage = [&](int buf, int kb) {
    // A tile: 128 x 32 bf16 already in memory -> async DMA straight to LDS
    const bf16_t* ga = A + (size_t)(bm + srow) * K + kb * 32 + scol;
    const unsigned la = lds_off(&sA[buf][srow * GLDA + scol]);
    async_cp16(la,      ga);
    async_cp16(la + 16, (const char*)ga + 16);
    // W tile: 128 x 32 fp32 -> bf16 on the fly (register path)
    const float* gw = W + (size_t)(bn + srow) * K + kb * 32 + scol;
    __builtin_prefetch((const char*)gw + 256, 0, 3);  // K-slab two ahead (kb+2) of this row
    v4f w0 = *(const v4f*)gw;
    v4f w1 = *(const v4f*)(gw + 4);
    v4f w2 = *(const v4f*)(gw + 8);
    v4f w3 = *(const v4f*)(gw + 12);
    v8bf b0, b1;
#pragma unroll
    for (int i = 0; i < 4; ++i) {
      b0[i]     = (bf16_t)w0[i];
      b0[i + 4] = (bf16_t)w1[i];
      b1[i]     = (bf16_t)w2[i];
      b1[i + 4] = (bf16_t)w3[i];
    }
    *(v8bf*)&sB[buf][srow * GLDA + scol]     = b0;
    *(v8bf*)&sB[buf][srow * GLDA + scol + 8] = b1;
  };

  stage(0, 0);
  wait_async0();
  __syncthreads();

  for (int kb = 0; kb < KB; ++kb) {
    const int buf = kb & 1;
    if (kb + 1 < KB) stage(buf ^ 1, kb + 1);

    v16bf Af[2], Bf[4];
#pragma unroll
    for (int mt = 0; mt < 2; ++mt)
      Af[mt] = ld_frag(&sA[buf][(wm + mt * 16 + fr) * GLDA + fk]);
#pragma unroll
    for (int nt = 0; nt < 4; ++nt)
      Bf[nt] = ld_frag(&sB[buf][(wn + nt * 16 + fr) * GLDA + fk]);
#pragma unroll
    for (int mt = 0; mt < 2; ++mt)
#pragma unroll
      for (int nt = 0; nt < 4; ++nt)
        acc[mt][nt] = __builtin_amdgcn_wmma_f32_16x16x32_bf16(
            false, Af[mt], false, Bf[nt], (short)0, acc[mt][nt], false, false);
    wait_async0();
    __syncthreads();
  }

#pragma unroll
  for (int mt = 0; mt < 2; ++mt)
#pragma unroll
    for (int nt = 0; nt < 4; ++nt) {
      const int n = bn + wn + nt * 16 + fr;
#pragma unroll
      for (int e = 0; e < 8; ++e) {
        const int m = bm + wm + mt * 16 + fk + e;
        const size_t idx = (size_t)m * N + n;
        float val = acc[mt][nt][e];
        if (epi == 1) val += RES[idx];
        C[idx] = val;
      }
    }
}

// ---------------------------------------------------------------- RoPE + pack to bf16 (Q,K head-major; V transposed)
__global__ __launch_bounds__(128) void rope_pack_kernel(
    const float* __restrict__ qf, const float* __restrict__ kf,
    const float* __restrict__ vf, const int* __restrict__ curpos,
    bf16_t* __restrict__ qb, bf16_t* __restrict__ kb, bf16_t* __restrict__ vt)
{
  const int S = 2048, HD = 128, H = 32, G = 8;
  const int s = blockIdx.x;
  const int j = blockIdx.y;  // 0..H-1: q, H..H+G-1: k, rest: v-transpose
  const int t = threadIdx.x;
  const float pos = (float)(curpos[0] + s);

  if (j < H) {
    if (t < 64) {
      const float* src = qf + ((size_t)s * H + j) * HD;
      const float x1 = src[2 * t], x2 = src[2 * t + 1];
      const float fr = __powf(10000.f, -(float)(2 * t) / (float)HD);
      float sn, cs;
      __sincosf(pos * fr, &sn, &cs);
      bf16_t* dst = qb + ((size_t)j * S + s) * HD;
      dst[2 * t]     = (bf16_t)(x1 * cs - x2 * sn);
      dst[2 * t + 1] = (bf16_t)(x1 * sn + x2 * cs);
    }
  } else if (j < H + G) {
    if (t < 64) {
      const int gh = j - H;
      const float* src = kf + ((size_t)s * G + gh) * HD;
      const float x1 = src[2 * t], x2 = src[2 * t + 1];
      const float fr = __powf(10000.f, -(float)(2 * t) / (float)HD);
      float sn, cs;
      __sincosf(pos * fr, &sn, &cs);
      bf16_t* dst = kb + ((size_t)gh * S + s) * HD;
      dst[2 * t]     = (bf16_t)(x1 * cs - x2 * sn);
      dst[2 * t + 1] = (bf16_t)(x1 * sn + x2 * cs);
    }
  } else {
    const int gh = j - H - G;
    const float* src = vf + ((size_t)s * G + gh) * HD;
    vt[((size_t)gh * HD + t) * S + s] = (bf16_t)src[t];
  }
}

// ---------------------------------------------------------------- Flash-style causal attention (GQA 4:1)
#define ALD 136  // halves per LDS row (128 data + 8 pad)

__global__ __launch_bounds__(256) void attn_kernel(
    const bf16_t* __restrict__ Qb, const bf16_t* __restrict__ Kb,
    const bf16_t* __restrict__ Vt, bf16_t* __restrict__ Ob,
    const int* __restrict__ curpos)
{
  const int S = 2048, HD = 128, H = 32;
  __shared__ bf16_t sQ[128 * ALD];
  __shared__ bf16_t sK[128 * ALD];
  __shared__ bf16_t sV[128 * ALD];   // V^T tile: [hd][key]
  __shared__ bf16_t sP[128 * ALD];   // probs, bf16
  __shared__ float    rowm[128], rowl[128], rowa[128];
  __shared__ unsigned tmaxU[128];

  const int qblk = blockIdx.x;
  const int h    = blockIdx.y;
  const int g    = h >> 2;  // H/G == 4
  const int t    = threadIdx.x;
  const int lane = t & 31;
  const int wid  = t >> 5;
  const int wm   = (wid & 3) * 32;
  const int wn   = (wid >> 2) * 64;
  const int fr   = lane & 15;
  const int fk   = (lane >> 4) * 8;
  const int cp   = curpos[0];

  const int srow = t >> 1;
  const int scol = (t & 1) * 64;  // halves

  {  // stage Q tile once (async DMA to LDS)
    const bf16_t* src = Qb + ((size_t)h * S + qblk * 128 + srow) * HD + scol;
    const unsigned lq = lds_off(&sQ[srow * ALD + scol]);
#pragma unroll
    for (int c = 0; c < 8; ++c)
      async_cp16(lq + c * 16, (const char*)src + c * 16);
  }
  if (t < 128) { rowm[t] = -3.0e38f; rowl[t] = 0.f; }

  v8f oacc[2][4];
#pragma unroll
  for (int i = 0; i < 2; ++i)
#pragma unroll
    for (int j = 0; j < 4; ++j) oacc[i][j] = (v8f)0.f;

  const int qlast = cp + qblk * 128 + 127;  // max allowed kpos for last row
  int jmax = qlast >> 7;
  if (jmax > S / 128 - 1) jmax = S / 128 - 1;
  const float rscale = 0.08838834764831845f;  // 1/sqrt(128)

  __syncthreads();

  for (int j = 0; j <= jmax; ++j) {
    {  // stage K tile and V^T tile (async DMA to LDS)
      const bf16_t* ks = Kb + ((size_t)g * S + j * 128 + srow) * HD + scol;
      const unsigned lk = lds_off(&sK[srow * ALD + scol]);
      const bf16_t* vs = Vt + ((size_t)g * HD + srow) * S + j * 128 + scol;
      const unsigned lv = lds_off(&sV[srow * ALD + scol]);
#pragma unroll
      for (int c = 0; c < 8; ++c) {
        async_cp16(lk + c * 16, (const char*)ks + c * 16);
        async_cp16(lv + c * 16, (const char*)vs + c * 16);
      }
    }
    if (t < 128) tmaxU[t] = f2ord(rowm[t]);
    wait_async0();
    __syncthreads();

    // ---- S = Q K^T (scaled)
    v8f sacc[2][4];
#pragma unroll
    for (int i = 0; i < 2; ++i)
#pragma unroll
      for (int jj = 0; jj < 4; ++jj) sacc[i][jj] = (v8f)0.f;
#pragma unroll
    for (int kc = 0; kc < 4; ++kc) {
      v16bf Af[2], Bf[4];
#pragma unroll
      for (int mt = 0; mt < 2; ++mt)
        Af[mt] = ld_frag(&sQ[(wm + mt * 16 + fr) * ALD + kc * 32 + fk]);
#pragma unroll
      for (int nt = 0; nt < 4; ++nt)
        Bf[nt] = ld_frag(&sK[(wn + nt * 16 + fr) * ALD + kc * 32 + fk]);
#pragma unroll
      for (int mt = 0; mt < 2; ++mt)
#pragma unroll
        for (int nt = 0; nt < 4; ++nt)
          sacc[mt][nt] = __builtin_amdgcn_wmma_f32_16x16x32_bf16(
              false, Af[mt], false, Bf[nt], (short)0, sacc[mt][nt], false, false);
    }

    // ---- scale, causal mask, row-max (ordered-uint ds_max atomics)
    const bool edge = (j * 128 + 127) > (cp + qblk * 128);
#pragma unroll
    for (int mt = 0; mt < 2; ++mt)
#pragma unroll
      for (int e = 0; e < 8; ++e) {
        const int m = wm + mt * 16 + fk + e;
        float rmax = -3.0e38f;
#pragma unroll
        for (int nt = 0; nt < 4; ++nt) {
          const int n = wn + nt * 16 + fr;
          float v = sacc[mt][nt][e] * rscale;
          if (edge && (j * 128 + n) > (cp + qblk * 128 + m)) v = -1.0e30f;
          sacc[mt][nt][e] = v;
          rmax = fmaxf(rmax, v);
        }
        atomicMax(&tmaxU[m], f2ord(rmax));
      }
    __syncthreads();

    if (t < 128) {
      const float mnew = ord2f(tmaxU[t]);
      const float a    = __expf(rowm[t] - mnew);
      rowa[t] = a;
      rowm[t] = mnew;
      rowl[t] *= a;
    }
    __syncthreads();

    // ---- P = exp(S - m), store bf16; rescale O accumulators
#pragma unroll
    for (int mt = 0; mt < 2; ++mt)
#pragma unroll
      for (int e = 0; e < 8; ++e) {
        const int m = wm + mt * 16 + fk + e;
        const float mr = rowm[m];
        const float al = rowa[m];
#pragma unroll
        for (int nt = 0; nt < 4; ++nt) {
          const int n = wn + nt * 16 + fr;
          const float p = __expf(sacc[mt][nt][e] - mr);
          sP[m * ALD + n] = (bf16_t)p;
          oacc[mt][nt][e] *= al;
        }
      }
    __syncthreads();

    if (t < 128) {  // row sums of P
      float ssum = 0.f;
      for (int c = 0; c < 128; ++c) ssum += (float)sP[t * ALD + c];
      rowl[t] += ssum;
    }

    // ---- O += P V  (V^T staged, so B-frags are contiguous)
#pragma unroll
    for (int kc = 0; kc < 4; ++kc) {
      v16bf Af[2], Bf[4];
#pragma unroll
      for (int mt = 0; mt < 2; ++mt)
        Af[mt] = ld_frag(&sP[(wm + mt * 16 + fr) * ALD + kc * 32 + fk]);
#pragma unroll
      for (int nt = 0; nt < 4; ++nt)
        Bf[nt] = ld_frag(&sV[(wn + nt * 16 + fr) * ALD + kc * 32 + fk]);
#pragma unroll
      for (int mt = 0; mt < 2; ++mt)
#pragma unroll
        for (int nt = 0; nt < 4; ++nt)
          oacc[mt][nt] = __builtin_amdgcn_wmma_f32_16x16x32_bf16(
              false, Af[mt], false, Bf[nt], (short)0, oacc[mt][nt], false, false);
    }
    __syncthreads();
  }

  if (t < 128) rowa[t] = 1.f / rowl[t];
  __syncthreads();

#pragma unroll
  for (int mt = 0; mt < 2; ++mt)
#pragma unroll
    for (int e = 0; e < 8; ++e) {
      const int m = wm + mt * 16 + fk + e;
      const float inv = rowa[m];
#pragma unroll
      for (int nt = 0; nt < 4; ++nt) {
        const int n = wn + nt * 16 + fr;
        Ob[(size_t)(qblk * 128 + m) * (H * HD) + h * HD + n] =
            (bf16_t)(oacc[mt][nt][e] * inv);
      }
    }
}

// ---------------------------------------------------------------- silu(a1) * a3 -> bf16
__global__ __launch_bounds__(256) void silu_mul_kernel(
    const float* __restrict__ a1, const float* __restrict__ a3,
    bf16_t* __restrict__ out, size_t n)
{
  const size_t i = (size_t)blockIdx.x * 256 + threadIdx.x;
  if (i < n) {
    const float x = a1[i];
    const float sig = 1.f / (1.f + __expf(-x));
    out[i] = (bf16_t)(x * sig * a3[i]);
  }
}

// ---------------------------------------------------------------- host launch
extern "C" void kernel_launch(void* const* d_in, const int* in_sizes, int n_in,
                              void* d_out, int out_size, void* d_ws, size_t ws_size,
                              hipStream_t stream)
{
  (void)in_sizes; (void)n_in; (void)out_size; (void)ws_size;
  const int S = 2048, D = 4096, F = 11008, H = 32, G = 8, HD = 128;

  const float* x      = (const float*)d_in[0];
  const float* wq     = (const float*)d_in[1];
  const float* wk     = (const float*)d_in[2];
  const float* wv     = (const float*)d_in[3];
  const float* wo     = (const float*)d_in[4];
  const float* w1     = (const float*)d_in[5];
  const float* w2     = (const float*)d_in[6];
  const float* w3     = (const float*)d_in[7];
  const float* g_attn = (const float*)d_in[8];
  const float* g_ffn  = (const float*)d_in[9];
  const int*   curpos = (const int*)d_in[10];

  char* p = (char*)d_ws;
  auto alloc = [&](size_t b) -> void* {
    void* r = (void*)p;
    p += (b + 255) & ~(size_t)255;
    return r;
  };
  bf16_t* hb    = (bf16_t*)alloc((size_t)S * D * 2);
  float*  qf    = (float*) alloc((size_t)S * H * HD * 4);
  float*  kf    = (float*) alloc((size_t)S * G * HD * 4);
  float*  vf    = (float*) alloc((size_t)S * G * HD * 4);
  bf16_t* qb    = (bf16_t*)alloc((size_t)S * H * HD * 2);
  bf16_t* kb    = (bf16_t*)alloc((size_t)S * G * HD * 2);
  bf16_t* vtb   = (bf16_t*)alloc((size_t)S * G * HD * 2);
  bf16_t* ob    = (bf16_t*)alloc((size_t)S * D * 2);
  float*  xattn = (float*) alloc((size_t)S * D * 4);
  bf16_t* h2b   = (bf16_t*)alloc((size_t)S * D * 2);
  float*  a1    = (float*) alloc((size_t)S * F * 4);
  float*  a3    = (float*) alloc((size_t)S * F * 4);
  bf16_t* ffb   = (bf16_t*)alloc((size_t)S * F * 2);

  const dim3 blk(256);

  rmsnorm_kernel<<<S, 256, 0, stream>>>(x, g_attn, hb, D);
  gemm_bf16_kernel<<<dim3(D / 128, S / 128), blk, 0, stream>>>(hb, wq, nullptr, qf, S, D, D, 0);
  gemm_bf16_kernel<<<dim3(G * HD / 128, S / 128), blk, 0, stream>>>(hb, wk, nullptr, kf, S, G * HD, D, 0);
  gemm_bf16_kernel<<<dim3(G * HD / 128, S / 128), blk, 0, stream>>>(hb, wv, nullptr, vf, S, G * HD, D, 0);
  rope_pack_kernel<<<dim3(S, H + 2 * G), 128, 0, stream>>>(qf, kf, vf, curpos, qb, kb, vtb);
  attn_kernel<<<dim3(S / 128, H), blk, 0, stream>>>(qb, kb, vtb, ob, curpos);
  gemm_bf16_kernel<<<dim3(D / 128, S / 128), blk, 0, stream>>>(ob, wo, x, xattn, S, D, D, 1);
  rmsnorm_kernel<<<S, 256, 0, stream>>>(xattn, g_ffn, h2b, D);
  gemm_bf16_kernel<<<dim3(F / 128, S / 128), blk, 0, stream>>>(h2b, w1, nullptr, a1, S, F, D, 0);
  gemm_bf16_kernel<<<dim3(F / 128, S / 128), blk, 0, stream>>>(h2b, w3, nullptr, a3, S, F, D, 0);
  silu_mul_kernel<<<dim3((unsigned)((size_t)S * F / 256)), blk, 0, stream>>>(a1, a3, ffb, (size_t)S * F);
  gemm_bf16_kernel<<<dim3(D / 128, S / 128), blk, 0, stream>>>(ffb, w2, xattn, (float*)d_out, S, D, F, 1);
}
